// NeuralSDE_64347200028967
// MI455X (gfx1250) — matrix operand order repaired and checked by reading, not compiled, not run on hardware
//
#include <hip/hip_runtime.h>

// Problem constants (from reference): S=128, H=512, K=16, B=2048, T=67
#define S_DIM 128
#define H_DIM 512
#define K_DIM 16
#define B_DIM 2048
#define NSTEP 63      // T-4 integration steps
#define K1PAD 160     // (S+1)=129 padded up to a multiple of 32 for the K loop

typedef __attribute__((ext_vector_type(16))) __bf16 v16bf;
typedef __attribute__((ext_vector_type(8)))  float  v8f;

union Frag { v16bf v; uint4 q[2]; };

__device__ __forceinline__ unsigned short f2bf(float x) {
    union { float f; unsigned int u; } v; v.f = x;
    unsigned int r = v.u + 0x7FFFu + ((v.u >> 16) & 1u);   // round-to-nearest-even
    return (unsigned short)(r >> 16);
}

// CDNA5 async global->LDS copy, 16B per lane (tracked by ASYNCcnt, no VGPRs).
// LDS destination address = low 32 bits of the generic shared-aperture address.
__device__ __forceinline__ void async_ld_b128(void* lds_ptr, const void* gptr) {
    unsigned laddr = (unsigned)(unsigned long long)lds_ptr;
    asm volatile("global_load_async_to_lds_b128 %0, %1, off"
                 :: "v"(laddr), "v"(gptr) : "memory");
}
__device__ __forceinline__ void wait_async3() {
    asm volatile("s_wait_asynccnt 0x3" ::: "memory");
}
__device__ __forceinline__ void wait_async0() {
    asm volatile("s_wait_asynccnt 0x0" ::: "memory");
}

// ---------------------------------------------------------------------------
// Pack fp32 weights (Krows x N, row-major) into the CDNA5 WMMA bf16 B-fragment
// layout, zero-padded to KT*32 rows. Element ((nt*KT+kt)*32+lane)*16+e holds
// W[kt*32 + 8*(lane>>4) + (e&7) + 16*(e>>3)][nt*16 + (lane&15)], so each lane's fragment is
// 32 contiguous bytes.
// ---------------------------------------------------------------------------
__global__ void pack_b_kernel(const float* __restrict__ W, unsigned short* __restrict__ Bp,
                              int Krows, int N, int KT) {
    int idx = blockIdx.x * blockDim.x + threadIdx.x;
    int total = KT * 32 * N;
    if (idx >= total) return;
    int e    = idx & 15;
    int lane = (idx >> 4) & 31;
    int rest = idx >> 9;
    int kt   = rest % KT;
    int nt   = rest / KT;
    int k = kt * 32 + ((lane >> 4) << 3) + (e & 7) + ((e >> 3) << 4);
    int n = (nt << 4) + (lane & 15);
    float val = (k < Krows) ? W[(size_t)k * N + n] : 0.0f;
    Bp[idx] = f2bf(val);
}

// Encoder layer 1 (K=3, once): h = relu(x @ enc_w1 + enc_b1) in bf16.
__global__ void enc1_kernel(const float* __restrict__ x, const float* __restrict__ w,
                            const float* __restrict__ b, unsigned short* __restrict__ h) {
    int idx = blockIdx.x * blockDim.x + threadIdx.x;
    if (idx >= B_DIM * H_DIM) return;
    int bi = idx / H_DIM, j = idx % H_DIM;
    float v = b[j];
    v = fmaf(x[bi * 3 + 0], w[0 * H_DIM + j], v);
    v = fmaf(x[bi * 3 + 1], w[1 * H_DIM + j], v);
    v = fmaf(x[bi * 3 + 2], w[2 * H_DIM + j], v);
    h[idx] = f2bf(fmaxf(v, 0.0f));
}

// zin = [t | y (+gp)] as (B, K1PAD) bf16, zero padded.
__global__ void make_zin_kernel(unsigned short* __restrict__ zin, const float* __restrict__ ts,
                                int step, const float* __restrict__ y, const float* __restrict__ gp) {
    int idx = blockIdx.x * blockDim.x + threadIdx.x;
    if (idx >= B_DIM * K1PAD) return;
    int b = idx / K1PAD, j = idx % K1PAD;
    float val = 0.0f;
    if (j == 0) {
        val = ts[3 + step];
    } else if (j <= S_DIM) {
        val = y[(size_t)b * S_DIM + (j - 1)];
        if (gp) val += gp[(size_t)b * S_DIM + (j - 1)];
    }
    zin[idx] = f2bf(val);
}

// y_{i+1} = y + drift*dt + 0.5*(gp + gp')
__global__ void update_y_kernel(float* __restrict__ yout, const float* __restrict__ yin,
                                const float* __restrict__ drift, const float* __restrict__ gp,
                                const float* __restrict__ gp2, const float* __restrict__ ts, int step) {
    int idx = blockIdx.x * blockDim.x + threadIdx.x;
    if (idx >= B_DIM * S_DIM) return;
    float dt = ts[4 + step] - ts[3 + step];
    yout[idx] = yin[idx] + drift[idx] * dt + 0.5f * (gp[idx] + gp2[idx]);
}

// ---------------------------------------------------------------------------
// Async-pipelined bf16 WMMA GEMM. Block tile 128x64, 256 threads (8 waves as
// 4x2), wave tile 32x32 = 2x2 v_wmma_f32_16x16x32_bf16, K-step 32.
// Per K-step the whole block stages A (128x32, 8KB) and B (4 n-tiles, 4KB)
// into a double-buffered LDS region via global_load_async_to_lds_b128
// (3 async ops/thread/stage), overlapping the next stage's copies with the
// current stage's WMMAs; fragments are then read as contiguous ds_load_b128.
// MODE: 0 = relu -> bf16, 1 = plain -> f32, 2 = fused einsum('bsk,bk->bs') -> f32.
// ---------------------------------------------------------------------------
template<int MODE>
__global__ __launch_bounds__(256) void gemm_stage_wmma(
    const unsigned short* __restrict__ A, int lda,
    const unsigned short* __restrict__ Bp,
    const float* __restrict__ bias,
    void* __restrict__ Cout, int N, int K,
    const float* __restrict__ dW, const float* __restrict__ ts, int step)
{
    __shared__ __align__(16) unsigned short sA[2 * 128 * 32];  // 16 KB
    __shared__ __align__(16) unsigned short sB[2 * 4 * 512];   //  8 KB

    const int tid   = threadIdx.x;
    const int lane  = tid & 31;
    const int wid   = tid >> 5;
    const int waveM = wid & 3, waveN = wid >> 2;
    const int h  = lane >> 4, nl = lane & 15;
    const int mBlk  = blockIdx.x * 128;
    const int nBase = blockIdx.y * 64 + waveN * 32;
    const int ntB4  = blockIdx.y * 4;          // first global n-tile of block
    const int KT = K >> 5;

    // staging indices (per thread, fixed)
    const int sArow = tid >> 2, sAcc = (tid & 3) * 8;          // 2 iters: +64 rows
    const int sBnt  = tid >> 6, sBrem = (tid & 63) * 8;
    const unsigned short* gA0 = A + (size_t)(mBlk + sArow) * lda + sAcc;
    const unsigned short* gA1 = A + (size_t)(mBlk + sArow + 64) * lda + sAcc;
    const unsigned short* gB  = Bp + ((size_t)(ntB4 + sBnt) * KT) * 512 + sBrem;

    // fragment read offsets (per lane, fixed)
    const int aoff0 = (waveM * 32 + nl) * 32 + h * 8;
    const int aoff1 = (waveM * 32 + 16 + nl) * 32 + h * 8;
    const int boff0 = (waveN * 2) * 512 + lane * 16;
    const int boff1 = (waveN * 2 + 1) * 512 + lane * 16;

    v8f acc[2][2];
#pragma unroll
    for (int i = 0; i < 2; ++i)
#pragma unroll
        for (int j = 0; j < 2; ++j)
#pragma unroll
            for (int v = 0; v < 8; ++v) acc[i][j][v] = 0.0f;

    // prologue: stage k=0 into buffer 0
    async_ld_b128(&sA[sArow * 32 + sAcc],        gA0);
    async_ld_b128(&sA[(sArow + 64) * 32 + sAcc], gA1);
    async_ld_b128(&sB[sBnt * 512 + sBrem],       gB);

    for (int kt = 0; kt < KT; ++kt) {
        const int cur = kt & 1, nxt = cur ^ 1;
        if (kt + 1 < KT) {   // stage k+1 into the other buffer
            const int ko = (kt + 1) * 32;
            async_ld_b128(&sA[nxt * 4096 + sArow * 32 + sAcc],        gA0 + ko);
            async_ld_b128(&sA[nxt * 4096 + (sArow + 64) * 32 + sAcc], gA1 + ko);
            async_ld_b128(&sB[nxt * 2048 + sBnt * 512 + sBrem],       gB + (size_t)(kt + 1) * 512);
            wait_async3();   // stage k done for this wave (in-order completion)
        } else {
            wait_async0();
        }
        __syncthreads();     // stage k visible block-wide

        Frag a0, a1, b0, b1;
        const unsigned short* sa = sA + cur * 4096;
        const unsigned short* sb = sB + cur * 2048;
        a0.q[0] = *(const uint4*)(sa + aoff0);
        a0.q[1] = *(const uint4*)(sa + aoff0 + 16);
        a1.q[0] = *(const uint4*)(sa + aoff1);
        a1.q[1] = *(const uint4*)(sa + aoff1 + 16);
        b0.q[0] = *(const uint4*)(sb + boff0);
        b0.q[1] = *(const uint4*)(sb + boff0 + 8);
        b1.q[0] = *(const uint4*)(sb + boff1);
        b1.q[1] = *(const uint4*)(sb + boff1 + 8);

        acc[0][0] = __builtin_amdgcn_wmma_f32_16x16x32_bf16(false, a0.v, false, b0.v, (short)0, acc[0][0], false, false);
        acc[0][1] = __builtin_amdgcn_wmma_f32_16x16x32_bf16(false, a0.v, false, b1.v, (short)0, acc[0][1], false, false);
        acc[1][0] = __builtin_amdgcn_wmma_f32_16x16x32_bf16(false, a1.v, false, b0.v, (short)0, acc[1][0], false, false);
        acc[1][1] = __builtin_amdgcn_wmma_f32_16x16x32_bf16(false, a1.v, false, b1.v, (short)0, acc[1][1], false, false);

        __syncthreads();     // all reads of buffer k done before it is re-staged
    }

    // Epilogue. C layout: lane (h, nl), VGPR v -> row = base + v + 8h, col = nl.
    const int mBase = mBlk + waveM * 32;
    if constexpr (MODE != 2) {
#pragma unroll
        for (int nj = 0; nj < 2; ++nj) {
            int col = nBase + nj * 16 + nl;
            float bv = bias[col];
#pragma unroll
            for (int mi = 0; mi < 2; ++mi) {
                int rbase = mBase + mi * 16 + h * 8;
                v8f c = acc[mi][nj];
#pragma unroll
                for (int v = 0; v < 8; ++v) {
                    float val = c[v] + bv;
                    if (MODE == 0) val = fmaxf(val, 0.0f);
                    size_t off = (size_t)(rbase + v) * N + col;
                    if (MODE == 1) ((float*)Cout)[off] = val;
                    else ((unsigned short*)Cout)[off] = f2bf(val);
                }
            }
        }
    } else {
        // Fused diffusion epilogue: each 16-wide n-tile is one s (K_DIM==16);
        // gp[b,s] = sum_k (C[b,s*16+k]+g_b2) * dW[step,b,k] * sqrt(dt) via a
        // 16-lane butterfly per half-wave; g (2048x2048) never hits memory.
        float sdt = sqrtf(ts[4 + step] - ts[3 + step]);
#pragma unroll
        for (int nj = 0; nj < 2; ++nj) {
            int col = nBase + nj * 16 + nl;  // col = s*16 + k, k = nl
            int s = col >> 4;
            float bv = bias[col];
#pragma unroll
            for (int mi = 0; mi < 2; ++mi) {
                int rbase = mBase + mi * 16 + h * 8;
                v8f c = acc[mi][nj];
                float p[8];
#pragma unroll
                for (int v = 0; v < 8; ++v)
                    p[v] = (c[v] + bv) * dW[((size_t)step * B_DIM + (rbase + v)) * K_DIM + nl];
#pragma unroll
                for (int off = 1; off < 16; off <<= 1)
#pragma unroll
                    for (int v = 0; v < 8; ++v)
                        p[v] += __shfl_xor(p[v], off, 32);
                if (nl == 0) {
#pragma unroll
                    for (int v = 0; v < 8; ++v)
                        ((float*)Cout)[(size_t)(rbase + v) * S_DIM + s] = p[v] * sdt;
                }
            }
        }
    }
}

// ---------------------------------------------------------------------------
extern "C" void kernel_launch(void* const* d_in, const int* in_sizes, int n_in,
                              void* d_out, int out_size, void* d_ws, size_t ws_size,
                              hipStream_t stream) {
    const float* ts     = (const float*)d_in[0];
    const float* x      = (const float*)d_in[1];
    const float* enc_w1 = (const float*)d_in[2];
    const float* enc_b1 = (const float*)d_in[3];
    const float* enc_w2 = (const float*)d_in[4];
    const float* enc_b2 = (const float*)d_in[5];
    const float* f_w1   = (const float*)d_in[6];
    const float* f_b1   = (const float*)d_in[7];
    const float* f_w2   = (const float*)d_in[8];
    const float* f_b2   = (const float*)d_in[9];
    const float* f_w3   = (const float*)d_in[10];
    const float* f_b3   = (const float*)d_in[11];
    const float* g_w1   = (const float*)d_in[12];
    const float* g_b1   = (const float*)d_in[13];
    const float* g_w2   = (const float*)d_in[14];
    const float* g_b2   = (const float*)d_in[15];
    const float* dW     = (const float*)d_in[16];
    float* out = (float*)d_out;

    // Workspace carve-up (~11 MB, 256B aligned)
    char* ws = (char*)d_ws;
    size_t off = 0;
    auto carve = [&](size_t bytes) -> char* {
        char* p = ws + off;
        off = (off + bytes + 255) & ~(size_t)255;
        return p;
    };
    unsigned short* enc_w2p = (unsigned short*)carve((size_t)512   * 128  * 2);
    unsigned short* f_w1p   = (unsigned short*)carve((size_t)K1PAD * 512  * 2);
    unsigned short* f_w2p   = (unsigned short*)carve((size_t)512   * 512  * 2);
    unsigned short* f_w3p   = (unsigned short*)carve((size_t)512   * 128  * 2);
    unsigned short* g_w1p   = (unsigned short*)carve((size_t)K1PAD * 512  * 2);
    unsigned short* g_w2p   = (unsigned short*)carve((size_t)512   * 2048 * 2);
    unsigned short* hbuf1   = (unsigned short*)carve((size_t)B_DIM * 512  * 2);
    unsigned short* hbuf2   = (unsigned short*)carve((size_t)B_DIM * 512  * 2);
    unsigned short* zin     = (unsigned short*)carve((size_t)B_DIM * K1PAD * 2);
    unsigned short* zin2    = (unsigned short*)carve((size_t)B_DIM * K1PAD * 2);
    float* drift = (float*)carve((size_t)B_DIM * S_DIM * 4);
    float* gp    = (float*)carve((size_t)B_DIM * S_DIM * 4);
    float* gp2   = (float*)carve((size_t)B_DIM * S_DIM * 4);

    const int TPB = 256;
    auto blocks = [](int n) { return (n + 255) / 256; };
    const float* nofp = nullptr;

    // --- one-time: pack weights to bf16 B-fragment layout ---
    pack_b_kernel<<<blocks(512 * 128),   TPB, 0, stream>>>(enc_w2, enc_w2p, 512, 128,  512 / 32);
    pack_b_kernel<<<blocks(K1PAD * 512), TPB, 0, stream>>>(f_w1,   f_w1p,   129, 512,  K1PAD / 32);
    pack_b_kernel<<<blocks(512 * 512),   TPB, 0, stream>>>(f_w2,   f_w2p,   512, 512,  512 / 32);
    pack_b_kernel<<<blocks(512 * 128),   TPB, 0, stream>>>(f_w3,   f_w3p,   512, 128,  512 / 32);
    pack_b_kernel<<<blocks(K1PAD * 512), TPB, 0, stream>>>(g_w1,   g_w1p,   129, 512,  K1PAD / 32);
    pack_b_kernel<<<blocks(512 * 2048),  TPB, 0, stream>>>(g_w2,   g_w2p,   512, 2048, 512 / 32);

    // --- y0 = relu(x@enc_w1+b1) @ enc_w2 + b2, straight into d_out slab 0 ---
    enc1_kernel<<<blocks(B_DIM * H_DIM), TPB, 0, stream>>>(x, enc_w1, enc_b1, hbuf1);
    gemm_stage_wmma<1><<<dim3(B_DIM / 128, 128 / 64), TPB, 0, stream>>>(
        hbuf1, 512, enc_w2p, enc_b2, out, 128, 512, nofp, nofp, 0);

    // --- 63 sequential SDE steps; y state lives in d_out slabs ---
    for (int i = 0; i < NSTEP; ++i) {
        const float* yin = out + (size_t)i * B_DIM * S_DIM;
        float* yout      = out + (size_t)(i + 1) * B_DIM * S_DIM;

        make_zin_kernel<<<blocks(B_DIM * K1PAD), TPB, 0, stream>>>(zin, ts, i, yin, nofp);

        // drift MLP
        gemm_stage_wmma<0><<<dim3(B_DIM / 128, 512 / 64), TPB, 0, stream>>>(
            zin, K1PAD, f_w1p, f_b1, hbuf1, 512, K1PAD, nofp, nofp, 0);
        gemm_stage_wmma<0><<<dim3(B_DIM / 128, 512 / 64), TPB, 0, stream>>>(
            hbuf1, 512, f_w2p, f_b2, hbuf2, 512, 512, nofp, nofp, 0);
        gemm_stage_wmma<1><<<dim3(B_DIM / 128, 128 / 64), TPB, 0, stream>>>(
            hbuf2, 512, f_w3p, f_b3, drift, 128, 512, nofp, nofp, 0);

        // gp = g_prod(t, y, dw)
        gemm_stage_wmma<0><<<dim3(B_DIM / 128, 512 / 64), TPB, 0, stream>>>(
            zin, K1PAD, g_w1p, g_b1, hbuf1, 512, K1PAD, nofp, nofp, 0);
        gemm_stage_wmma<2><<<dim3(B_DIM / 128, 2048 / 64), TPB, 0, stream>>>(
            hbuf1, 512, g_w2p, g_b2, gp, 2048, 512, dW, ts, i);

        // gp' = g_prod(t, y + gp, dw)
        make_zin_kernel<<<blocks(B_DIM * K1PAD), TPB, 0, stream>>>(zin2, ts, i, yin, gp);
        gemm_stage_wmma<0><<<dim3(B_DIM / 128, 512 / 64), TPB, 0, stream>>>(
            zin2, K1PAD, g_w1p, g_b1, hbuf1, 512, K1PAD, nofp, nofp, 0);
        gemm_stage_wmma<2><<<dim3(B_DIM / 128, 2048 / 64), TPB, 0, stream>>>(
            hbuf1, 512, g_w2p, g_b2, gp2, 2048, 512, dW, ts, i);

        update_y_kernel<<<blocks(B_DIM * S_DIM), TPB, 0, stream>>>(yout, yin, drift, gp, gp2, ts, i);
    }
}
